// GNN3_31061203485250
// MI455X (gfx1250) — matrix operand order repaired and checked by reading, not compiled, not run on hardware
//
#include <hip/hip_runtime.h>
#include <hip/hip_bf16.h>
#include <math.h>

// ---------------------------------------------------------------------------
// GNN3: GraphNorm + 3xGCN + GlobalAttention pooling + MLP head, fp32 with
// f16 WMMA for all 128x128 GEMMs.  N=100000, E=1600000, B=256, H=128.
// GEMM epilogue fuses the GCN self-loop + bias init of the aggregation
// buffer (saves a full 100MB read+write pass per layer).
// ---------------------------------------------------------------------------

#define HDIM 128
#define NGRAPH 256
#define EPS 1e-5f

typedef __attribute__((ext_vector_type(16))) _Float16 v16h;
typedef __attribute__((ext_vector_type(8)))  float    v8f;

// ---------------------------------------------------------------------------
// Degree / normalization helpers
// ---------------------------------------------------------------------------
__global__ void deg_init_k(float* deg, int N) {
    int i = blockIdx.x * blockDim.x + threadIdx.x;
    if (i < N) deg[i] = 1.0f;                       // self-loop contributes 1
}

__global__ void deg_acc_k(const long long* __restrict__ ei, float* deg, int E) {
    int i = blockIdx.x * blockDim.x + threadIdx.x;
    if (i < E) atomicAdd(&deg[(int)ei[(long long)E + i]], 1.0f);   // dst row
}

__global__ void dinv_k(float* deg, int N) {
    int i = blockIdx.x * blockDim.x + threadIdx.x;
    if (i < N) deg[i] = rsqrtf(fmaxf(deg[i], 1.0f));   // deg -> d^{-1/2}
}

// ---------------------------------------------------------------------------
// Segment start offsets from sorted batch vector (B graphs, starts[B] = N)
// ---------------------------------------------------------------------------
__global__ void starts_init_k(int* starts, int N) {
    int i = blockIdx.x * blockDim.x + threadIdx.x;
    if (i <= NGRAPH) starts[i] = N;
}

__global__ void starts_mark_k(const long long* __restrict__ batch, int* starts, int N) {
    int i = blockIdx.x * blockDim.x + threadIdx.x;
    if (i >= N) return;
    int b = (int)batch[i];
    if (i == 0 || (int)batch[i - 1] != b) atomicMin(&starts[b], i);
}

__global__ void starts_fix_k(int* starts) {
    if (threadIdx.x == 0 && blockIdx.x == 0) {
        for (int g = NGRAPH - 1; g >= 0; --g)
            if (starts[g] > starts[g + 1]) starts[g] = starts[g + 1]; // empty graphs
    }
}

// ---------------------------------------------------------------------------
// GraphNorm: out = w * (x - mean_g*ms) / sqrt(var_g + eps) + b   (per graph)
// One workgroup per graph; batch is sorted so segments are contiguous.
// ---------------------------------------------------------------------------
template<int C, bool RELU>
__global__ __launch_bounds__(256)
void graphnorm_k(const float* __restrict__ X, float* __restrict__ Y,
                 const int* __restrict__ starts,
                 const float* __restrict__ w, const float* __restrict__ b,
                 const float* __restrict__ ms) {
    constexpr int SLOTS = 256 / C;
    __shared__ float red[256];
    __shared__ float smean[C];
    __shared__ float sistd[C];

    int g = blockIdx.x;
    int s = starts[g], e = starts[g + 1];
    float cnt = fmaxf((float)(e - s), 1.0f);
    int t = threadIdx.x;
    int c = t % C, slot = t / C;

    // pass 1: per-channel sums
    float acc = 0.0f;
    for (int i = s + slot; i < e; i += SLOTS) acc += X[i * C + c];
    red[t] = acc; __syncthreads();
    for (int h = SLOTS / 2; h > 0; h >>= 1) {
        if (slot < h) red[t] += red[t + h * C];
        __syncthreads();
    }
    if (t < C) smean[t] = red[t] / cnt;
    __syncthreads();
    float mn = smean[c] * ms[c];

    // pass 2: variance of centered values
    acc = 0.0f;
    for (int i = s + slot; i < e; i += SLOTS) {
        float d = X[i * C + c] - mn; acc += d * d;
    }
    red[t] = acc; __syncthreads();
    for (int h = SLOTS / 2; h > 0; h >>= 1) {
        if (slot < h) red[t] += red[t + h * C];
        __syncthreads();
    }
    if (t < C) sistd[t] = rsqrtf(red[t] / cnt + EPS);
    __syncthreads();

    float wc = w[c], bc = b[c], is = sistd[c];
    for (int i = s + slot; i < e; i += SLOTS) {
        float v = wc * (X[i * C + c] - mn) * is + bc;
        if (RELU) v = fmaxf(v, 0.0f);
        Y[i * C + c] = v;
    }
}

// ---------------------------------------------------------------------------
// X[M,4] @ W[4,128] -> Y[M,128]; also fused GCN-aggregation init:
// Agg[n,c] = bias[c] + dinv[n]^2 * Y[n,c]
// ---------------------------------------------------------------------------
__global__ void gemm4_k(const float* __restrict__ X, const float* __restrict__ W,
                        const float* __restrict__ dinv, const float* __restrict__ bias,
                        float* __restrict__ Y, float* __restrict__ Agg, int M) {
    int gid = blockIdx.x * blockDim.x + threadIdx.x;
    if (gid >= M * HDIM) return;
    int n = gid >> 7, c = gid & 127;
    const float* xr = X + n * 4;
    float v = xr[0] * W[c] + xr[1] * W[HDIM + c] + xr[2] * W[2 * HDIM + c]
            + xr[3] * W[3 * HDIM + c];
    Y[gid] = v;
    float dv = dinv[n];
    Agg[gid] = bias[c] + dv * dv * v;
}

// ---------------------------------------------------------------------------
// WMMA GEMM: Y[M,128] = X[M,128] @ W[128,128]
//   !HAS_AGG: Y = (relu?)(X@W + bias)
//   HAS_AGG : Y = X@W ; Agg = aggBias + dinv^2 * Y   (GCN self-loop init)
// Block = 256 threads (8 waves), 128 rows. A-tile in LDS as f16 (stride 136
// halves), dinv^2 staged in LDS. Each wave owns one 16-col tile; B fragments
// live in registers, reused across 8 row tiles. A and B share a contiguous k order (k = 16*hi + e).
// All hot-path indexing is 32-bit; epilogue flags are compile-time.
// ---------------------------------------------------------------------------
template<bool RELU, bool HAS_AGG>
__global__ __launch_bounds__(256)
void gemm128_wmma_k(const float* __restrict__ X, const float* __restrict__ W,
                    const float* __restrict__ bias, float* __restrict__ Y,
                    const float* __restrict__ dinv, const float* __restrict__ aggBias,
                    float* __restrict__ Agg, int M) {
    __shared__ _Float16 sA[128 * 136];
    __shared__ float sD2[128];
    int t = threadIdx.x;
    int wave = t >> 5, lane = t & 31;
    int row0 = blockIdx.x * 128;
    int nn = lane & 15;          // column within 16-wide tile / row M within A tile
    int hi = lane >> 4;          // lane-half selector
    int n0 = wave * 16;          // this wave's output column tile

    // B fragments (held across the whole block's row tiles): 4 k-steps of 32.
    // Layout: element e of v16h holds W[k][n] with k = ks*32 + hi*16 + e.
    v16h bfrag[4];
    #pragma unroll
    for (int ks = 0; ks < 4; ++ks) {
        int kb = ks * 32 + hi * 16;
        #pragma unroll
        for (int e = 0; e < 16; ++e)
            bfrag[ks][e] = (_Float16)W[(kb + e) * HDIM + n0 + nn];
    }

    // Stage A tile: 128x128 fp32 -> f16 in LDS (zero-pad past M).
    for (int i = t; i < 128 * 128; i += 256) {
        int r = i >> 7, c = i & 127;
        int gr = row0 + r;
        float v = (gr < M) ? X[gr * HDIM + c] : 0.0f;
        sA[r * 136 + c] = (_Float16)v;
    }
    if (HAS_AGG && t < 128) {
        int gr = row0 + t;
        float dv = (gr < M) ? dinv[gr] : 0.0f;
        sD2[t] = dv * dv;
    }
    __syncthreads();

    int col = n0 + nn;
    float bval  = RELU    ? bias[col]    : 0.0f;
    float abval = HAS_AGG ? aggBias[col] : 0.0f;
    bool full = (row0 + 128 <= M);

    if (full) {
        #pragma unroll
        for (int rt = 0; rt < 8; ++rt) {
            v8f acc = {};
            const _Float16* arow = &sA[(rt * 16 + nn) * 136];
            #pragma unroll
            for (int ks = 0; ks < 4; ++ks) {
                v16h afrag;
                int kb0 = ks * 32 + hi * 16;
                #pragma unroll
                for (int j = 0; j < 8; ++j) {
                    int koff = kb0 + 2 * j;
                    afrag[2 * j]     = arow[koff];
                    afrag[2 * j + 1] = arow[koff + 1];
                }
                acc = __builtin_amdgcn_wmma_f32_16x16x32_f16(
                    false, afrag, false, bfrag[ks], (short)0, acc, false, false);
            }
            int mloc = rt * 16 + hi * 8;               // row within block tile
            #pragma unroll
            for (int r = 0; r < 8; ++r) {
                int idx = (row0 + mloc + r) * HDIM + col;   // 32-bit index
                float v = acc[r];
                if (RELU) v = fmaxf(v + bval, 0.0f);
                Y[idx] = v;
                if (HAS_AGG) Agg[idx] = abval + sD2[mloc + r] * v;
            }
        }
    } else {
        #pragma unroll
        for (int rt = 0; rt < 8; ++rt) {
            v8f acc = {};
            const _Float16* arow = &sA[(rt * 16 + nn) * 136];
            #pragma unroll
            for (int ks = 0; ks < 4; ++ks) {
                v16h afrag;
                int kb0 = ks * 32 + hi * 16;
                #pragma unroll
                for (int j = 0; j < 8; ++j) {
                    int koff = kb0 + 2 * j;
                    afrag[2 * j]     = arow[koff];
                    afrag[2 * j + 1] = arow[koff + 1];
                }
                acc = __builtin_amdgcn_wmma_f32_16x16x32_f16(
                    false, afrag, false, bfrag[ks], (short)0, acc, false, false);
            }
            int mloc = rt * 16 + hi * 8;
            #pragma unroll
            for (int r = 0; r < 8; ++r) {
                int grow = row0 + mloc + r;
                if (grow < M) {
                    int idx = grow * HDIM + col;
                    float v = acc[r];
                    if (RELU) v = fmaxf(v + bval, 0.0f);
                    Y[idx] = v;
                    if (HAS_AGG) Agg[idx] = abval + sD2[mloc + r] * v;
                }
            }
        }
    }
}

// ---------------------------------------------------------------------------
// GCN edge aggregation: out[dst] += dinv[src]*dinv[dst] * T[src]
// (out already initialized with bias + self-loop term by the GEMM epilogue)
// ---------------------------------------------------------------------------
__global__ __launch_bounds__(256)
void scatter_edges_k(const float* __restrict__ T, const long long* __restrict__ ei,
                     const float* __restrict__ dinv, float* __restrict__ out, int E) {
    int gid = blockIdx.x * blockDim.x + threadIdx.x;
    int e = gid >> 5;
    if (e >= E) return;
    int lane = gid & 31;
    int s = (int)ei[e];                       // src
    int d = (int)ei[(long long)E + e];        // dst
    float nrm = dinv[s] * dinv[d];
    float4 v = ((const float4*)(T + s * HDIM))[lane];
    float* o = out + d * HDIM + lane * 4;
    atomicAdd(o + 0, v.x * nrm);
    atomicAdd(o + 1, v.y * nrm);
    atomicAdd(o + 2, v.z * nrm);
    atomicAdd(o + 3, v.w * nrm);
}

// ---------------------------------------------------------------------------
// Row dot with a 128-vector: out[m] = X[m,:].v + bias[0]  (one wave per row)
// ---------------------------------------------------------------------------
__global__ void dotcol_k(const float* __restrict__ X, const float* __restrict__ v,
                         const float* __restrict__ bias, float* __restrict__ out, int M) {
    int gid = blockIdx.x * blockDim.x + threadIdx.x;
    int row = gid >> 5;
    if (row >= M) return;
    int lane = gid & 31;
    float4 a = ((const float4*)(X + row * HDIM))[lane];
    float4 w = ((const float4*)v)[lane];
    float p = a.x * w.x + a.y * w.y + a.z * w.z + a.w * w.w;
    #pragma unroll
    for (int off = 16; off > 0; off >>= 1) p += __shfl_xor(p, off, 32);
    if (lane == 0) out[row] = p + bias[0];
}

// ---------------------------------------------------------------------------
// Per-graph segment softmax over gate + attention-weighted pooling of Hf.
// ---------------------------------------------------------------------------
__global__ __launch_bounds__(256)
void attnpool_k(const float* __restrict__ gate, const float* __restrict__ Hf,
                const int* __restrict__ starts, float* __restrict__ pooled) {
    __shared__ float red[256];
    __shared__ float sgmax, sden;
    int g = blockIdx.x;
    int s = starts[g], e = starts[g + 1];
    int t = threadIdx.x;

    float m = -3.4e38f;
    for (int i = s + t; i < e; i += 256) m = fmaxf(m, gate[i]);
    red[t] = m; __syncthreads();
    for (int h = 128; h > 0; h >>= 1) {
        if (t < h) red[t] = fmaxf(red[t], red[t + h]);
        __syncthreads();
    }
    if (t == 0) sgmax = red[0];
    __syncthreads();
    float gm = sgmax;

    float acc = 0.0f;
    for (int i = s + t; i < e; i += 256) acc += __expf(gate[i] - gm);
    red[t] = acc; __syncthreads();
    for (int h = 128; h > 0; h >>= 1) {
        if (t < h) red[t] += red[t + h];
        __syncthreads();
    }
    if (t == 0) sden = red[0];
    __syncthreads();
    float inv = (sden > 0.0f) ? (1.0f / sden) : 0.0f;

    int c = t & 127, slot = t >> 7;   // 2 node slots x 128 channels
    float pacc = 0.0f;
    for (int i = s + slot; i < e; i += 2)
        pacc += __expf(gate[i] - gm) * inv * Hf[i * HDIM + c];
    red[t] = pacc; __syncthreads();
    if (t < 128) pooled[g * HDIM + t] = (e > s) ? (red[t] + red[t + 128]) : 0.0f;
}

// ---------------------------------------------------------------------------
// Launch
// ---------------------------------------------------------------------------
extern "C" void kernel_launch(void* const* d_in, const int* in_sizes, int n_in,
                              void* d_out, int out_size, void* d_ws, size_t ws_size,
                              hipStream_t stream) {
    const float*     x     = (const float*)d_in[0];
    const long long* ei    = (const long long*)d_in[1];
    const long long* batch = (const long long*)d_in[2];
    const float *W1 = (const float*)d_in[3],  *b1 = (const float*)d_in[4];
    const float *W2 = (const float*)d_in[5],  *b2 = (const float*)d_in[6];
    const float *W3 = (const float*)d_in[7],  *b3 = (const float*)d_in[8];
    const float *gn0w = (const float*)d_in[9],  *gn0b = (const float*)d_in[10], *gn0m = (const float*)d_in[11];
    const float *gn1w = (const float*)d_in[12], *gn1b = (const float*)d_in[13], *gn1m = (const float*)d_in[14];
    const float *gn2w = (const float*)d_in[15], *gn2b = (const float*)d_in[16], *gn2m = (const float*)d_in[17];
    const float *gn3w = (const float*)d_in[18], *gn3b = (const float*)d_in[19], *gn3m = (const float*)d_in[20];
    const float *g1 = (const float*)d_in[21], *gb1 = (const float*)d_in[22];
    const float *g2 = (const float*)d_in[23], *gb2 = (const float*)d_in[24];
    const float *g3 = (const float*)d_in[25], *gb3 = (const float*)d_in[26];
    const float *l1 = (const float*)d_in[27], *lb1 = (const float*)d_in[28];
    const float *l2 = (const float*)d_in[29], *lb2 = (const float*)d_in[30];
    const float *l3 = (const float*)d_in[31], *lb3 = (const float*)d_in[32];
    float* out = (float*)d_out;

    const int N = in_sizes[0] / 4;
    const int E = in_sizes[1] / 2;

    // Workspace carve-up (floats)
    float* ws = (float*)d_ws;
    size_t off = 0;
    float* deg   = ws + off; off += (size_t)N;
    int*   starts = (int*)(ws + off); off += (size_t)(NGRAPH + 16);
    float* hn0   = ws + off; off += (size_t)N * 4;
    float* P     = ws + off; off += (size_t)N * HDIM;
    float* Q     = ws + off; off += (size_t)N * HDIM;
    float* R     = ws + off; off += (size_t)N * HDIM;
    float* gate  = ws + off; off += (size_t)N;
    float* pooled = ws + off; off += (size_t)NGRAPH * HDIM;
    float* mlp1  = ws + off; off += (size_t)NGRAPH * HDIM;
    float* mlp2  = ws + off; off += (size_t)NGRAPH * HDIM;

    const int T = 256;
    dim3 blk(T);
    int gN   = (N + T - 1) / T;
    int gE   = (E + T - 1) / T;
    int gNH  = (N * HDIM + T - 1) / T;          // elementwise over N*128
    int gEw  = (int)(((long long)E * 32 + T - 1) / T); // wave per edge
    int gNw  = (N * 32 + T - 1) / T;             // wave per node
    int gGemm = (N + 127) / 128;
    int gGemmB = (NGRAPH + 127) / 128;

    // degrees -> d^{-1/2}
    deg_init_k<<<gN, blk, 0, stream>>>(deg, N);
    deg_acc_k<<<gE, blk, 0, stream>>>(ei, deg, E);
    dinv_k<<<gN, blk, 0, stream>>>(deg, N);

    // graph segment boundaries
    starts_init_k<<<2, blk, 0, stream>>>(starts, N);
    starts_mark_k<<<gN, blk, 0, stream>>>(batch, starts, N);
    starts_fix_k<<<1, 32, 0, stream>>>(starts);

    // GraphNorm(x) -> hn0 [N,4]
    graphnorm_k<4, false><<<NGRAPH, blk, 0, stream>>>(x, hn0, starts, gn0w, gn0b, gn0m);

    // layer 1: hn0 @ W1 -> P with fused Q-init ; edge scatter -> Q ; gn+relu -> P
    gemm4_k<<<gNH, blk, 0, stream>>>(hn0, W1, deg, b1, P, Q, N);
    scatter_edges_k<<<gEw, blk, 0, stream>>>(P, ei, deg, Q, E);
    graphnorm_k<HDIM, true><<<NGRAPH, blk, 0, stream>>>(Q, P, starts, gn1w, gn1b, gn1m);

    // layer 2: P @ W2 -> Q with fused agg init in R ; scatter -> R ; gn+relu -> P
    gemm128_wmma_k<false, true><<<gGemm, blk, 0, stream>>>(P, W2, nullptr, Q, deg, b2, R, N);
    scatter_edges_k<<<gEw, blk, 0, stream>>>(Q, ei, deg, R, E);
    graphnorm_k<HDIM, true><<<NGRAPH, blk, 0, stream>>>(R, P, starts, gn2w, gn2b, gn2m);

    // layer 3: P @ W3 -> Q with fused agg init in R ; scatter -> R ; gn -> P
    gemm128_wmma_k<false, true><<<gGemm, blk, 0, stream>>>(P, W3, nullptr, Q, deg, b3, R, N);
    scatter_edges_k<<<gEw, blk, 0, stream>>>(Q, ei, deg, R, E);
    graphnorm_k<HDIM, false><<<NGRAPH, blk, 0, stream>>>(R, P, starts, gn3w, gn3b, gn3m);
    // P now holds final node features h

    // gate MLP: relu(h@g1+gb1) -> Q ; relu(Q@g2+gb2) -> R ; R.g3+gb3 -> gate
    gemm128_wmma_k<true, false><<<gGemm, blk, 0, stream>>>(P, g1, gb1, Q, nullptr, nullptr, nullptr, N);
    gemm128_wmma_k<true, false><<<gGemm, blk, 0, stream>>>(Q, g2, gb2, R, nullptr, nullptr, nullptr, N);
    dotcol_k<<<gNw, blk, 0, stream>>>(R, g3, gb3, gate, N);

    // segment softmax + attention pooling -> pooled [B,128]
    attnpool_k<<<NGRAPH, blk, 0, stream>>>(gate, P, starts, pooled);

    // head MLP
    gemm128_wmma_k<true, false><<<gGemmB, blk, 0, stream>>>(pooled, l1, lb1, mlp1, nullptr, nullptr, nullptr, NGRAPH);
    gemm128_wmma_k<true, false><<<gGemmB, blk, 0, stream>>>(mlp1, l2, lb2, mlp2, nullptr, nullptr, nullptr, NGRAPH);
    dotcol_k<<<(NGRAPH * 32 + T - 1) / T, blk, 0, stream>>>(mlp2, l3, lb3, out, NGRAPH);
}
